// DCSF_act_27161373180093
// MI455X (gfx1250) — compile-verified, matrix operand order checked by reading, not run on hardware
//
#include <hip/hip_runtime.h>
#include <hip/hip_bf16.h>

// ---------------------------------------------------------------------------
// DCSF model on MI455X (gfx1250): all 1x1 convs + dense layers lowered to a
// single generic bf16 WMMA GEMM kernel (v_wmma_f32_16x16x32_bf16, fp32 acc),
// layer-by-layer through L2-resident workspace buffers.
// Tile staging uses GLOBAL_LOAD_ASYNC_TO_LDS_B128 (ASYNCcnt) when available.
// ---------------------------------------------------------------------------

#define B_   32
#define Q_   16
#define T_   512
#define DV_  17           // values channel dim
#define NSEG (B_ * Q_)    // 512 segment rows
#define RR   ((long)NSEG * T_)   // 262144 resnet rows
#define BT   ((long)B_ * T_)     // 16384 dense rows

typedef __attribute__((ext_vector_type(16))) __bf16 v16bf;
typedef __attribute__((ext_vector_type(8)))  float  v8f;

union ABu { v16bf v; unsigned u[8]; };

#if defined(__has_builtin)
# if __has_builtin(__builtin_amdgcn_global_load_async_to_lds_b128) && \
     __has_builtin(__builtin_amdgcn_s_wait_asynccnt)
#  define USE_ASYNC_LDS 1
# endif
#endif
#ifndef USE_ASYNC_LDS
# define USE_ASYNC_LDS 0
#endif

#if USE_ASYNC_LDS
// Builtin signature (from hipcc diagnostic): param0 is
//   'int __attribute__((vector_size(16))) __device__ *'  (AS1, global)
// param1 is the LDS (AS3) pointer of the same vector type.
typedef int v4i_ld __attribute__((vector_size(16)));
typedef __attribute__((address_space(1))) v4i_ld* g128_t;
typedef __attribute__((address_space(3))) v4i_ld* l128_t;
#endif

__device__ __forceinline__ float bf2f(unsigned short u) {
    union { float f; unsigned u32; } c;
    c.u32 = ((unsigned)u) << 16;
    return c.f;
}
__device__ __forceinline__ unsigned short f2bf(float f) {
    union { float f; unsigned u; } c;
    c.f = f;
    unsigned r = c.u + 0x7FFFu + ((c.u >> 16) & 1u);   // round to nearest even
    return (unsigned short)(r >> 16);
}

// ---------------------------------------------------------------------------
// Prep: X0[r, 0..31] = bf16( concat(values, tt)[r] * m[r] ), zero-pad 18..31
// ---------------------------------------------------------------------------
__global__ void prep_kernel(const float* __restrict__ values,
                            const float* __restrict__ tt,
                            const float* __restrict__ mask,
                            unsigned short* __restrict__ X0) {
    long i = (long)blockIdx.x * blockDim.x + threadIdx.x;   // over RR*32
    if (i >= RR * 32) return;
    long r = i >> 5;
    int  k = (int)(i & 31);
    float v = 0.f;
    if (k < DV_)        v = values[r * DV_ + k];
    else if (k == DV_)  v = tt[r];
    v *= mask[r];
    X0[i] = f2bf(v);
}

// ---------------------------------------------------------------------------
// Weight prep (once per matrix): Wt[coutp][cinp] = bf16(W[cin][cout]^T),
// zero padded so GEMM staging needs no bounds checks and rows are 16B-aligned.
// ---------------------------------------------------------------------------
__global__ void wprep_kernel(const float* __restrict__ W, int cin, int cout,
                             int cinp, int coutp, unsigned short* __restrict__ Wt) {
    long i = (long)blockIdx.x * blockDim.x + threadIdx.x;   // coutp*cinp
    if (i >= (long)coutp * cinp) return;
    int n = (int)(i / cinp), k = (int)(i - (long)n * cinp);
    unsigned short v = 0;
    if (k < cin && n < cout) v = f2bf(W[(long)k * cout + n]);
    Wt[i] = v;
}

// ---------------------------------------------------------------------------
// ids[n,t] = cumsum_t( m[n, times[n,t]] )   (literal reference gather index)
// ---------------------------------------------------------------------------
__global__ void ids_kernel(const float* __restrict__ mask,
                           const int* __restrict__ times,
                           int* __restrict__ ids) {
    int n = blockIdx.x * blockDim.x + threadIdx.x;
    if (n >= NSEG) return;
    const long base = (long)n * T_;
    int run = 0;
    for (int t = 0; t < T_; ++t) {
        int tm = times[base + t];
        run += (mask[base + tm] > 0.5f) ? 1 : 0;
        ids[base + t] = run;
    }
}

// ---------------------------------------------------------------------------
// Gather ("hold last valid" via ids) + masked segment sum over Q channels.
// ---------------------------------------------------------------------------
__global__ void gather_agg_kernel(const unsigned short* __restrict__ H,  // [RR,128] bf16
                                  const int* __restrict__ ids,
                                  const int* __restrict__ lengths,
                                  unsigned short* __restrict__ AGG) {    // [BT,128] bf16
    long i = (long)blockIdx.x * blockDim.x + threadIdx.x;  // BT*128
    if (i >= BT * 128) return;
    int  c  = (int)(i & 127);
    long bt = i >> 7;
    int  t  = (int)(bt % T_);
    int  b  = (int)(bt / T_);
    int  L  = lengths[b]; if (L > Q_) L = Q_;
    float s = 0.f;
    for (int q = 0; q < L; ++q) {
        long n  = (long)b * Q_ + q;
        int  id = ids[n * T_ + t];
        if (id > 0) s += bf2f(H[(n * T_ + (id - 1)) * 128 + c]);
    }
    AGG[bt * 128 + c] = f2bf(s);
}

// ---------------------------------------------------------------------------
// Generic WMMA GEMM:  Y[r,:] = act( (X[r,:cin] @ Wt^T + bias)
//                                   (* mask[r])? (+ Res[r,:])? )
// X/Res/Y bf16 workspace; Wt bf16 pre-transposed [coutp][cinp]; bias fp32.
// 64 threads = 2 wave32; 32 rows per workgroup; A tile + 16-wide W slice in
// LDS, staged via global_load_async_to_lds_b128 when available.
// act: 0=none, 1=relu, 2=sigmoid.  Yf32 != nullptr => fp32 output (stride cout).
// ---------------------------------------------------------------------------
#define KS 520   // LDS row stride in bf16 elems (1040B: 16B-aligned, bank skew)

__global__ __launch_bounds__(64)
void gemm_kernel(const unsigned short* __restrict__ X, int sx,
                 const unsigned short* __restrict__ Wt,
                 const float* __restrict__ bias,
                 int cin, int cout,
                 const unsigned short* __restrict__ Res, int sr,
                 const float* __restrict__ mask,
                 unsigned short* __restrict__ Y, int sy,
                 float* __restrict__ Yf32,
                 int act) {
    __shared__ unsigned short sA[32 * KS];
    __shared__ unsigned short sB[16 * KS];
    __shared__ float sBias[16];

    const int tid  = threadIdx.x;
    const int lane = tid & 31;
    const int wave = tid >> 5;
    const int hi   = lane >> 4;     // 0: K base, 1: K base + 8
    const int ln   = lane & 15;
    const long rowBase = (long)blockIdx.x * 32;
    const int cinp   = (cin + 31) & ~31;   // == sx for every call site
    const int perRow = cinp >> 3;          // 16B chunks per LDS/global row

    // ---- Stage A tile (32 rows x cinp bf16) into LDS.
#if USE_ASYNC_LDS
    for (int c = tid; c < 32 * perRow; c += 64) {
        int r = c / perRow, j = c - r * perRow;
        const unsigned short* g = X + (rowBase + r) * sx + (j << 3);
        __builtin_amdgcn_global_load_async_to_lds_b128(
            (g128_t)g, (l128_t)&sA[r * KS + (j << 3)], 0, 0);
    }
    __builtin_amdgcn_s_wait_asynccnt(0);
#else
    for (int idx = tid; idx < 32 * cinp; idx += 64) {
        int r = idx / cinp, k = idx - r * cinp;
        sA[r * KS + k] = X[(rowBase + r) * sx + k];
    }
#endif
    __syncthreads();

    const int ntiles = (cout + 15) >> 4;
    const int ktiles = cinp >> 5;
    const int arow   = wave * 16 + ln;

    for (int nt = 0; nt < ntiles; ++nt) {
        // ---- Stage 16 pre-transposed weight rows: sB[n][k] = Wt[nt*16+n][k]
#if USE_ASYNC_LDS
        for (int c = tid; c < 16 * perRow; c += 64) {
            int n = c / perRow, j = c - n * perRow;
            const unsigned short* g = Wt + (long)(nt * 16 + n) * cinp + (j << 3);
            __builtin_amdgcn_global_load_async_to_lds_b128(
                (g128_t)g, (l128_t)&sB[n * KS + (j << 3)], 0, 0);
        }
        __builtin_amdgcn_s_wait_asynccnt(0);
#else
        for (int idx = tid; idx < 16 * cinp; idx += 64) {
            int n = idx / cinp, k = idx - n * cinp;
            sB[n * KS + k] = Wt[(long)(nt * 16 + n) * cinp + k];
        }
#endif
        if (tid < 16) {
            int gn = nt * 16 + tid;
            sBias[tid] = (gn < cout) ? bias[gn] : 0.f;
        }
        __syncthreads();

        v8f acc = {0.f, 0.f, 0.f, 0.f, 0.f, 0.f, 0.f, 0.f};
        for (int kt = 0; kt < ktiles; ++kt) {
            ABu a, b;
#pragma unroll
            for (int p = 0; p < 8; ++p) {
                // 16-bit A/B fragment K map: VGPR p holds K pair
                //   base = (p<4 ? 2p : 16+2(p-4)) + (lane>=16 ? 8 : 0)
                int k = kt * 32 + ((p & 4) ? 16 : 0) + ((p & 3) << 1) + (hi << 3);
                a.u[p] = *(const unsigned*)&sA[arow * KS + k];
                b.u[p] = *(const unsigned*)&sB[ln * KS + k];
            }
            acc = __builtin_amdgcn_wmma_f32_16x16x32_bf16(
                false, a.v, false, b.v, (short)0, acc, false, false);
        }

        // ---- Epilogue: bias -> mask -> residual -> activation -> store.
        const float bn = sBias[ln];
        const int   gn = nt * 16 + ln;
#pragma unroll
        for (int i = 0; i < 8; ++i) {
            int  m    = i + (hi << 3);
            long grow = rowBase + wave * 16 + m;
            float v = acc[i] + bn;
            if (mask) v *= mask[grow];
            if (Res)  v += bf2f(Res[grow * (long)sr + gn]);
            if (act == 1)      v = v > 0.f ? v : 0.f;
            else if (act == 2) v = 1.f / (1.f + __expf(-v));
            if (gn < cout) {
                if (Yf32) Yf32[grow * (long)cout + gn] = v;
                else      Y[grow * (long)sy + gn] = f2bf(v);
            }
        }
        __syncthreads();
    }
}

// ---------------------------------------------------------------------------
extern "C" void kernel_launch(void* const* d_in, const int* in_sizes, int n_in,
                              void* d_out, int out_size, void* d_ws, size_t ws_size,
                              hipStream_t stream) {
    (void)in_sizes; (void)n_in; (void)out_size; (void)ws_size;

    // setup_inputs() order: values, tt, tim_pos,
    //   resnet_params[3 blocks x (wx,bx,wy,by,wz,bz,ws,bs)],
    //   dense_params[4 x (w,b)], times, lengths
    const float* values = (const float*)d_in[0];
    const float* tt     = (const float*)d_in[1];
    const float* mask   = (const float*)d_in[2];
    const float* rp[24];
    for (int i = 0; i < 24; ++i) rp[i] = (const float*)d_in[3 + i];
    const float* dW0 = (const float*)d_in[27]; const float* db0 = (const float*)d_in[28];
    const float* dW1 = (const float*)d_in[29]; const float* db1 = (const float*)d_in[30];
    const float* dW2 = (const float*)d_in[31]; const float* db2 = (const float*)d_in[32];
    const float* dW3 = (const float*)d_in[33]; const float* db3 = (const float*)d_in[34];
    const int* times   = (const int*)d_in[35];
    const int* lengths = (const int*)d_in[36];

    // Workspace layout (hand-aliased; total ~164 MB):
    //   [0,64MB)      B1 : RR x 128 bf16  (A1 = RR x 64 in [0,32MB),
    //                                      X0 = RR x 32 in [32,48MB), D2 reuse)
    //   [64,128MB)    B2 : RR x 128 bf16  (final resnet output h2)
    //   [128,160MB)   A2 : RR x 64 bf16   (h0; later AGG + D1 reuse)
    //   [160,161MB)   IDS: RR int32
    //   [162MB, ...)  Wt : 16 pre-transposed bf16 weight matrices (~1.5MB)
    const size_t MB = 1ull << 20;
    char* ws = (char*)d_ws;
    unsigned short* B1  = (unsigned short*)(ws);
    unsigned short* B2  = (unsigned short*)(ws + 64 * MB);
    unsigned short* A2  = (unsigned short*)(ws + 128 * MB);
    int*            IDS = (int*)(ws + 160 * MB);
    unsigned short* A1  = B1;                                   // dead before B1 use
    unsigned short* X0  = (unsigned short*)(ws + 32 * MB);      // dead before B1 full use
    unsigned short* AGG = A2;                                   // A2 dead post-resnet
    unsigned short* D1  = (unsigned short*)(ws + 128 * MB + 4 * MB);
    unsigned short* D2  = B1;                                   // B1 dead post-resnet

    // ---- Pre-transpose + convert all weight matrices to bf16 (Wt[coutp][cinp]).
    struct Mat { const float* W; int cin; int cout; };
    Mat mats[16] = {
        {rp[0],  DV_+1, 64}, {rp[2],  64, 64},   {rp[4],  64, 64},   {rp[6],  DV_+1, 64},
        {rp[8],  64, 128},   {rp[10], 128, 128}, {rp[12], 128, 128}, {rp[14], 64, 128},
        {rp[16], 128, 128},  {rp[18], 128, 128}, {rp[20], 128, 128}, {rp[22], 128, 128},
        {dW0, 128, 512},     {dW1, 512, 512},    {dW2, 512, 512},    {dW3, 512, 1}
    };
    unsigned short* wt[16];
    size_t off = 162 * MB;
    for (int i = 0; i < 16; ++i) {
        int cinp  = (mats[i].cin  + 31) & ~31;
        int coutp = (mats[i].cout + 15) & ~15;
        wt[i] = (unsigned short*)(ws + off);
        off = (off + (size_t)coutp * cinp * 2 + 255) & ~(size_t)255;
        long tot = (long)coutp * cinp;
        wprep_kernel<<<(unsigned)((tot + 255) / 256), 256, 0, stream>>>(
            mats[i].W, mats[i].cin, mats[i].cout, cinp, coutp, wt[i]);
    }

    {   long tot = RR * 32;
        prep_kernel<<<(unsigned)((tot + 255) / 256), 256, 0, stream>>>(values, tt, mask, X0); }
    ids_kernel<<<(NSEG + 255) / 256, 256, 0, stream>>>(mask, times, IDS);

    auto G = [&](const unsigned short* Xp, int sx, int wi, const float* bp,
                 const unsigned short* Rp, int sr,
                 const float* mk, unsigned short* Yp, int sy, float* Yf, int act, long rows) {
        gemm_kernel<<<(unsigned)(rows / 32), 64, 0, stream>>>(
            Xp, sx, wt[wi], bp, mats[wi].cin, mats[wi].cout, Rp, sr, mk, Yp, sy, Yf, act);
    };

    // ---- Resnet block 0 (18 -> 64): cx, cy, sc, cz+res (mask after each conv)
    G(X0, 32, 0,  rp[1],  nullptr, 0, mask, A1, 64, nullptr, 1, RR);  // cx
    G(A1, 64, 1,  rp[3],  nullptr, 0, mask, A1, 64, nullptr, 1, RR);  // cy (in-place)
    G(X0, 32, 3,  rp[7],  nullptr, 0, mask, A2, 64, nullptr, 0, RR);  // shortcut
    G(A1, 64, 2,  rp[5],  A2, 64,     mask, A2, 64, nullptr, 1, RR);  // cz+sc,relu -> h0

    // ---- Resnet block 1 (64 -> 128)
    G(A2, 64,  4, rp[9],  nullptr, 0, mask, B1, 128, nullptr, 1, RR); // cx
    G(B1, 128, 5, rp[11], nullptr, 0, mask, B1, 128, nullptr, 1, RR); // cy (in-place)
    G(A2, 64,  7, rp[15], nullptr, 0, mask, B2, 128, nullptr, 0, RR); // shortcut
    G(B1, 128, 6, rp[13], B2, 128,    mask, B2, 128, nullptr, 1, RR); // cz+sc -> h1

    // ---- Resnet block 2 (128 -> 128)
    G(B2, 128, 8,  rp[17], nullptr, 0, mask, B1, 128, nullptr, 1, RR); // cx
    G(B1, 128, 9,  rp[19], nullptr, 0, mask, B1, 128, nullptr, 1, RR); // cy (in-place)
    G(B2, 128, 11, rp[23], nullptr, 0, mask, B2, 128, nullptr, 0, RR); // shortcut (in-place)
    G(B1, 128, 10, rp[21], B2, 128,    mask, B2, 128, nullptr, 1, RR); // cz+sc -> h2

    // ---- Gather (hold-last-valid via ids) + segment sum over Q channels
    {   long tot = BT * 128;
        gather_agg_kernel<<<(unsigned)((tot + 255) / 256), 256, 0, stream>>>(B2, IDS, lengths, AGG); }

    // ---- Dense MLP: 128 -> 512 -> 512 -> 512 -> 1, sigmoid out (fp32)
    G(AGG, 128, 12, db0, nullptr, 0, nullptr, D1, 512, nullptr, 1, BT);
    G(D1,  512, 13, db1, nullptr, 0, nullptr, D2, 512, nullptr, 1, BT);
    G(D2,  512, 14, db2, nullptr, 0, nullptr, D1, 512, nullptr, 1, BT);
    G(D1,  512, 15, db3, nullptr, 0, nullptr, nullptr, 0, (float*)d_out, 2, BT);
}
